// FragJunctionAE_14620068675883
// MI455X (gfx1250) — matrix operand order repaired and checked by reading, not compiled, not run on hardware
//
#include <hip/hip_runtime.h>
#include <cstdint>
#include <cstddef>

// ---------------------------------------------------------------------------
// Types for WMMA bf16 (CDNA5 / gfx1250, wave32)
// ---------------------------------------------------------------------------
typedef __attribute__((ext_vector_type(16))) __bf16 v16bf;
typedef __attribute__((ext_vector_type(8)))  float  v8f;

union BFrag { v16bf v; unsigned int u[8]; };

__device__ __forceinline__ unsigned short f2bf(float f) {
  unsigned int u = __float_as_uint(f);
  u += 0x7fffu + ((u >> 16) & 1u);      // round-to-nearest-even
  return (unsigned short)(u >> 16);
}

// Up to 4 K-chunks of 128 columns each; chunk c supplies K in [128c,128c+128).
struct Chunks {
  const unsigned short* base[4];   // bf16 tables, row width = rowStride
  const int* idx[4];               // row gather index (null = identity)
  int colOff[4];                   // column offset within source row
  int rowStride[4];                // elements per source row
};

// ---------------------------------------------------------------------------
// Fused gather + GEMM:  Y[M x NO] = act( gatherA(M x K) @ Wt^T + bias [+res] )
// Wt is pre-transposed bf16 [NOpad x K].  Block: 256 threads = 8 waves.
// Tile: block covers 128 rows; wave w owns rows w*16..w*16+15, cols 32
// (two 16x16 WMMA tiles) selected by blockIdx.y.
// ---------------------------------------------------------------------------
__global__ void __launch_bounds__(256)
gemm_bf16_kernel(Chunks ch, const unsigned short* __restrict__ Wt,
                 const float* __restrict__ bias,
                 int M, int K, int NO, int relu,
                 const float* __restrict__ residual, int resStride,
                 float* __restrict__ outF, unsigned short* __restrict__ outB,
                 int outStride)
{
  __shared__ alignas(16) unsigned short As[128 * 32];

  const int tid  = threadIdx.x;
  const int lane = tid & 31;
  const int wave = tid >> 5;
  const int half = lane >> 4;
  const int l16  = lane & 15;
  const int mBlock = blockIdx.x * 128;
  const int nBase  = blockIdx.y * 32;

  v8f acc0 = {}; v8f acc1 = {};

  const int stRow  = tid >> 1;          // 0..127 : staged row
  const int stHalf = (tid & 1) * 16;    // 0 or 16 : element offset
  const int gr     = mBlock + stRow;    // global row

  for (int k = 0; k < K; k += 32) {
    // ---- stage A chunk (128 rows x 32 bf16) with fused gather ----
    {
      const int c = k >> 7;
      uint4 d0 = make_uint4(0u, 0u, 0u, 0u);
      uint4 d1 = make_uint4(0u, 0u, 0u, 0u);
      if (gr < M) {
        const int* ix = ch.idx[c];
        const int srcRow = ix ? ix[gr] : gr;
        const unsigned short* p = ch.base[c] +
            (size_t)srcRow * (size_t)ch.rowStride[c] + ch.colOff[c] + (k & 127) + stHalf;
        const uint4* p4 = reinterpret_cast<const uint4*>(p);
        d0 = p4[0]; d1 = p4[1];
      }
      uint4* s4 = reinterpret_cast<uint4*>(&As[stRow * 32 + stHalf]);
      s4[0] = d0; s4[1] = d1;
    }
    __syncthreads();

    // ---- A fragment (16x32 bf16, contiguous K: lane-half h holds K = 16h..16h+15) ----
    BFrag a;
    {
      const unsigned short* ap = &As[(wave * 16 + l16) * 32 + half * 16];
      const uint4 lo = *reinterpret_cast<const uint4*>(ap);        // K = h*16+0..7
      const uint4 hi = *reinterpret_cast<const uint4*>(ap + 8);    // K = h*16+8..15
      a.u[0] = lo.x; a.u[1] = lo.y; a.u[2] = lo.z; a.u[3] = lo.w;
      a.u[4] = hi.x; a.u[5] = hi.y; a.u[6] = hi.z; a.u[7] = hi.w;
    }

    // ---- B fragments (32x16 bf16: lane = column, 16 contiguous K / half) ----
    {
      const unsigned short* bp0 = Wt + (size_t)(nBase + l16) * (size_t)K + k + half * 16;
      __builtin_prefetch(bp0 + 32, 0, 1);
      const uint4 x0 = *reinterpret_cast<const uint4*>(bp0);
      const uint4 x1 = *reinterpret_cast<const uint4*>(bp0 + 8);
      BFrag b0;
      b0.u[0] = x0.x; b0.u[1] = x0.y; b0.u[2] = x0.z; b0.u[3] = x0.w;
      b0.u[4] = x1.x; b0.u[5] = x1.y; b0.u[6] = x1.z; b0.u[7] = x1.w;
      acc0 = __builtin_amdgcn_wmma_f32_16x16x32_bf16(false, a.v, false, b0.v,
                                                     (short)0, acc0, false, false);
      const unsigned short* bp1 = Wt + (size_t)(nBase + 16 + l16) * (size_t)K + k + half * 16;
      const uint4 y0 = *reinterpret_cast<const uint4*>(bp1);
      const uint4 y1 = *reinterpret_cast<const uint4*>(bp1 + 8);
      BFrag b1;
      b1.u[0] = y0.x; b1.u[1] = y0.y; b1.u[2] = y0.z; b1.u[3] = y0.w;
      b1.u[4] = y1.x; b1.u[5] = y1.y; b1.u[6] = y1.z; b1.u[7] = y1.w;
      acc1 = __builtin_amdgcn_wmma_f32_16x16x32_bf16(false, a.v, false, b1.v,
                                                     (short)0, acc1, false, false);
    }
    __syncthreads();
  }

  // ---- epilogue: D layout VGPR r -> (M=r | M=r+8) by lane half, N=lane&15 ----
#pragma unroll
  for (int t = 0; t < 2; ++t) {
    v8f acc = (t == 0) ? acc0 : acc1;
    const int col = nBase + t * 16 + l16;
    if (col < NO) {
      const float bv = bias ? bias[col] : 0.0f;
#pragma unroll
      for (int r = 0; r < 8; ++r) {
        const int row = mBlock + wave * 16 + r + half * 8;
        if (row < M) {
          float v = acc[r] + bv;
          if (residual) v += residual[(size_t)row * resStride + col];
          if (relu) v = v > 0.0f ? v : 0.0f;
          if (outF) outF[(size_t)row * outStride + col] = v;
          if (outB) outB[(size_t)row * outStride + col] = f2bf(v);
        }
      }
    }
  }
}

// ---------------------------------------------------------------------------
// LayerNorm over rows of width 128 (optional ReLU after), f32 in, f32/bf16 out
// ---------------------------------------------------------------------------
__global__ void __launch_bounds__(128)
ln128_kernel(const float* __restrict__ x, float* __restrict__ outF,
             unsigned short* __restrict__ outB, int M, int relu)
{
  __shared__ float sm[128];
  const int row = blockIdx.x;
  const int c = threadIdx.x;
  const float v = x[(size_t)row * 128 + c];
  sm[c] = v; __syncthreads();
  for (int s = 64; s > 0; s >>= 1) { if (c < s) sm[c] += sm[c + s]; __syncthreads(); }
  const float mean = sm[0] * (1.0f / 128.0f);
  __syncthreads();
  const float d = v - mean;
  sm[c] = d * d; __syncthreads();
  for (int s = 64; s > 0; s >>= 1) { if (c < s) sm[c] += sm[c + s]; __syncthreads(); }
  const float var = sm[0] * (1.0f / 128.0f);
  float y = d * rsqrtf(var + 1e-5f);
  if (relu) y = y > 0.0f ? y : 0.0f;
  if (outF) outF[(size_t)row * 128 + c] = y;
  if (outB) outB[(size_t)row * 128 + c] = f2bf(y);
}

// ---------------------------------------------------------------------------
// Weight prep: Wt[n][k] = bf16(W[k][n]), zero-padded to [NOpad x Kpad]
// ---------------------------------------------------------------------------
__global__ void prep_wt_kernel(const float* __restrict__ W, unsigned short* __restrict__ Wt,
                               int K, int Kpad, int NO, int NOpad)
{
  const int i = blockIdx.x * 256 + threadIdx.x;
  const int total = NOpad * Kpad;
  if (i >= total) return;
  const int n = i / Kpad, k = i % Kpad;
  const float v = (n < NO && k < K) ? W[(size_t)k * NO + n] : 0.0f;
  Wt[(size_t)n * Kpad + k] = f2bf(v);
}

// ---------------------------------------------------------------------------
// Node feature embedding: concat of four 32-wide embedding rows -> 128
// ---------------------------------------------------------------------------
__global__ void embed_kernel(const int* __restrict__ h, const int* __restrict__ jc,
                             const int* __restrict__ iff, const int* __restrict__ af,
                             const float* __restrict__ th, const float* __restrict__ tjc,
                             const float* __restrict__ tif, const float* __restrict__ taf,
                             float* __restrict__ outF, unsigned short* __restrict__ outB, int Nn)
{
  const int i = blockIdx.x * 256 + threadIdx.x;
  if (i >= Nn * 128) return;
  const int n = i >> 7, c = i & 127;
  const int g = c >> 5, cc = c & 31;
  float v;
  if (g == 0)      v = th [(size_t)h [n] * 32 + cc];
  else if (g == 1) v = tjc[(size_t)jc[n] * 32 + cc];
  else if (g == 2) { int x = iff[n]; x = x < 0 ? 0 : (x > 299 ? 299 : x); v = tif[(size_t)x * 32 + cc]; }
  else             { int x = af [n]; x = x < 0 ? 0 : (x > 29  ? 29  : x); v = taf[(size_t)x * 32 + cc]; }
  outF[i] = v; outB[i] = f2bf(v);
}

// Edge embedding (rows duplicated for bidi; first zeroFirst rows use type 0)
__global__ void edge_embed_kernel(const int* __restrict__ etype, const float* __restrict__ table,
                                  float* __restrict__ outF, unsigned short* __restrict__ outB,
                                  int Ed, int zeroFirst)
{
  const int i = blockIdx.x * 256 + threadIdx.x;
  if (i >= 2 * Ed * 128) return;
  const int r = i >> 7, c = i & 127;
  const int r0 = (r < Ed) ? r : r - Ed;
  const int t = (r0 < zeroFirst) ? 0 : etype[r0 - zeroFirst];
  const float v = table[(size_t)t * 128 + c];
  outF[i] = v; outB[i] = f2bf(v);
}

__global__ void build_edges_enc(const int* __restrict__ ei, const int* __restrict__ batch,
                                int* src, int* dst, int* bsrc, int E)
{
  const int i = blockIdx.x * 256 + threadIdx.x;
  if (i >= E) return;
  const int s = ei[i], d = ei[E + i];
  src[i] = s;     dst[i] = d;
  src[E + i] = d; dst[E + i] = s;
  bsrc[i] = batch[s]; bsrc[E + i] = batch[d];
}

__global__ void build_edges_dec(const int* __restrict__ ae, const int* __restrict__ de,
                                const int* __restrict__ batch,
                                int* src, int* dst, int* bsrc, int E3_, int E2_)
{
  const int Ed = E3_ + E2_;
  const int i = blockIdx.x * 256 + threadIdx.x;
  if (i >= Ed) return;
  const int s = (i < E3_) ? ae[i] : de[i - E3_];
  const int d = (i < E3_) ? ae[E3_ + i] : de[E2_ + (i - E3_)];
  src[i] = s;      dst[i] = d;
  src[Ed + i] = d; dst[Ed + i] = s;
  bsrc[i] = batch[s]; bsrc[Ed + i] = batch[d];
}

__global__ void count_kernel(const int* __restrict__ idx, float* cnt, int M)
{
  const int i = blockIdx.x * 256 + threadIdx.x;
  if (i < M) atomicAdd(&cnt[idx[i]], 1.0f);
}

// one thread handles 4 columns of one row (row = tid/32)
__global__ void seg_sum_kernel(const float* __restrict__ x, const int* __restrict__ idx,
                               float* sums, int M)
{
  const int t = blockIdx.x * 256 + threadIdx.x;
  const int row = t >> 5;
  if (row >= M) return;
  const int c = (t & 31) * 4;
  const int g = idx[row];
  const float4 v = *reinterpret_cast<const float4*>(x + (size_t)row * 128 + c);
  atomicAdd(&sums[(size_t)g * 128 + c + 0], v.x);
  atomicAdd(&sums[(size_t)g * 128 + c + 1], v.y);
  atomicAdd(&sums[(size_t)g * 128 + c + 2], v.z);
  atomicAdd(&sums[(size_t)g * 128 + c + 3], v.w);
}

__global__ void seg_div_kernel(const float* __restrict__ sums, const float* __restrict__ cnt,
                               float* outF, unsigned short* outB, int R)
{
  const int i = blockIdx.x * 256 + threadIdx.x;
  if (i >= R * 128) return;
  float c = cnt[i >> 7]; c = c < 1.0f ? 1.0f : c;
  const float v = sums[i] / c;
  if (outF) outF[i] = v;
  if (outB) outB[i] = f2bf(v);
}

// z = mu + eps*exp(0.5*logvar); also pack z into bf16 [B x 128] (cols 64+ zero)
__global__ void z_kernel(const float* __restrict__ eps, float* __restrict__ outAll,
                         unsigned short* __restrict__ zB, int BZ)
{
  const int i = blockIdx.x * 256 + threadIdx.x;
  if (i >= BZ) return;
  const float mu = outAll[i], lv = outAll[BZ + i];
  const float z = mu + eps[i] * expf(0.5f * lv);
  outAll[2 * BZ + i] = z;
  const int row = i >> 6, col = i & 63;
  zB[(size_t)row * 128 + col] = f2bf(z);
}

// ---------------------------------------------------------------------------
// Host orchestration
// ---------------------------------------------------------------------------
struct MpnnWts {
  unsigned short *e1, *e2, *g1, *g2, *h1, *h2;
  const float *e1b, *e2b, *g1b, *g2b, *h1b, *h2b;
};

extern "C" void kernel_launch(void* const* d_in, const int* in_sizes, int n_in,
                              void* d_out, int out_size, void* d_ws, size_t ws_size,
                              hipStream_t stream)
{
  (void)in_sizes; (void)n_in; (void)out_size; (void)ws_size;
  const int N = 200000, E1c = 150000, E2c = 150000, E3c = 100000, Bc = 8192, Zc = 64;
  const int EENC = 2 * E1c, EDHALF = E3c + E2c, EDEC = 2 * EDHALF;
  const int BZ = Bc * Zc;

  // ---- flattened input map (JAX pytree: dicts sorted alphabetically) ----
  const int*   ae    = (const int*)d_in[0];    // ae_to_pred_index [2,E3]
  const int*   batch = (const int*)d_in[1];
  const int*   dce   = (const int*)d_in[2];    // decomp_e
  const int*   dcei  = (const int*)d_in[3];    // decomp_e_index [2,E2]
  const int*   ein   = (const int*)d_in[4];    // e
  const int*   eidx  = (const int*)d_in[5];    // e_index [2,E1]
  const float* eps   = (const float*)d_in[6];
  const int*   hI    = (const int*)d_in[7];
  const int*   hafI  = (const int*)d_in[8];
  const int*   hifI  = (const int*)d_in[9];
  const int*   hjcI  = (const int*)d_in[10];
  // params leaves: 11..108 (see mapping in analysis)

  float* out = (float*)d_out;

  // ---- workspace bump allocator ----
  char* wsBase = (char*)d_ws; size_t wsOff = 0;
  auto alloc = [&](size_t bytes) -> void* {
    void* p = wsBase + wsOff; wsOff = (wsOff + bytes + 255) & ~(size_t)255; return p;
  };

  float*          hF  = (float*)alloc((size_t)N * 128 * 4);
  unsigned short* hB  = (unsigned short*)alloc((size_t)N * 128 * 2);
  float*          eF  = (float*)alloc((size_t)EDEC * 128 * 4);
  unsigned short* eB  = (unsigned short*)alloc((size_t)EDEC * 128 * 2);
  float*          gF  = (float*)alloc((size_t)Bc * 128 * 4);
  unsigned short* gB  = (unsigned short*)alloc((size_t)Bc * 128 * 2);
  unsigned short* mid = (unsigned short*)alloc((size_t)EDEC * 256 * 2);
  float*          tmpF= (float*)alloc((size_t)EDEC * 128 * 4);
  float*          aggF= (float*)alloc((size_t)N * 128 * 4);
  unsigned short* aggB= (unsigned short*)alloc((size_t)N * 128 * 2);
  float*          ghF = (float*)alloc((size_t)Bc * 128 * 4);
  unsigned short* ghB = (unsigned short*)alloc((size_t)Bc * 128 * 2);
  float*          gevF= (float*)alloc((size_t)Bc * 128 * 4);
  unsigned short* gevB= (unsigned short*)alloc((size_t)Bc * 128 * 2);
  float* nodeCnt  = (float*)alloc((size_t)Bc * 4);
  float* edgeCntE = (float*)alloc((size_t)Bc * 4);
  float* edgeCntD = (float*)alloc((size_t)Bc * 4);
  float* inCntE   = (float*)alloc((size_t)N * 4);
  float* inCntD   = (float*)alloc((size_t)N * 4);
  int* srcE = (int*)alloc((size_t)EENC * 4);
  int* dstE = (int*)alloc((size_t)EENC * 4);
  int* bsrcE= (int*)alloc((size_t)EENC * 4);
  int* srcD = (int*)alloc((size_t)EDEC * 4);
  int* dstD = (int*)alloc((size_t)EDEC * 4);
  int* bsrcD= (int*)alloc((size_t)EDEC * 4);
  unsigned short* zB = (unsigned short*)alloc((size_t)Bc * 128 * 2);

  // ---- helpers ----
  auto prep = [&](int leaf, int K, int NO, int Kpad, int NOpad, unsigned short* dst) {
    const int total = NOpad * Kpad;
    prep_wt_kernel<<<(total + 255) / 256, 256, 0, stream>>>(
        (const float*)d_in[leaf], dst, K, Kpad, NO, NOpad);
  };
  auto gemm = [&](const Chunks& ch, const unsigned short* Wt, const float* bias,
                  int M, int K, int NOpad, int NO, int relu,
                  const float* res, int resStride, float* oF, unsigned short* oB, int oStride) {
    dim3 grid((M + 127) / 128, NOpad / 32);
    gemm_bf16_kernel<<<grid, 256, 0, stream>>>(ch, Wt, bias, M, K, NO, relu,
                                               res, resStride, oF, oB, oStride);
  };
  auto ln = [&](const float* x, float* oF, unsigned short* oB, int M, int relu) {
    ln128_kernel<<<M, 128, 0, stream>>>(x, oF, oB, M, relu);
  };
  auto segsum = [&](const float* x, const int* idx, float* sums, int M) {
    const int tot = M * 32;
    seg_sum_kernel<<<(tot + 255) / 256, 256, 0, stream>>>(x, idx, sums, M);
  };
  auto segdiv = [&](const float* sums, const float* cnt, float* oF, unsigned short* oB, int R) {
    const int tot = R * 128;
    seg_div_kernel<<<(tot + 255) / 256, 256, 0, stream>>>(sums, cnt, oF, oB, R);
  };
  auto count = [&](const int* idx, float* cnt, int M) {
    count_kernel<<<(M + 255) / 256, 256, 0, stream>>>(idx, cnt, M);
  };
  auto zero = [&](void* p, size_t b) { hipMemsetAsync(p, 0, b, stream); };

  // ---- weight prep ----
  MpnnWts encW[3], decW[3];
  auto mkW = [&](MpnnWts& w, int base) {
    w.e1 = (unsigned short*)alloc(256 * 512 * 2);
    w.e2 = (unsigned short*)alloc(128 * 256 * 2);
    w.g1 = (unsigned short*)alloc(256 * 384 * 2);
    w.g2 = (unsigned short*)alloc(128 * 256 * 2);
    w.h1 = (unsigned short*)alloc(256 * 384 * 2);
    w.h2 = (unsigned short*)alloc(128 * 256 * 2);
    w.e1b = (const float*)d_in[base + 1];  w.e2b = (const float*)d_in[base + 3];
    w.g1b = (const float*)d_in[base + 5];  w.g2b = (const float*)d_in[base + 7];
    w.h1b = (const float*)d_in[base + 9];  w.h2b = (const float*)d_in[base + 11];
    prep(base + 0, 512, 256, 512, 256, w.e1);
    prep(base + 2, 256, 128, 256, 128, w.e2);
    prep(base + 4, 384, 256, 384, 256, w.g1);
    prep(base + 6, 256, 128, 256, 128, w.g2);
    prep(base + 8, 384, 256, 384, 256, w.h1);
    prep(base + 10, 256, 128, 256, 128, w.h2);
  };
  for (int l = 0; l < 3; ++l) { mkW(encW[l], 61 + 12 * l); mkW(decW[l], 16 + 12 * l); }

  unsigned short* muW1 = (unsigned short*)alloc(128 * 128 * 2);
  unsigned short* muW2 = (unsigned short*)alloc(64 * 128 * 2);
  unsigned short* lvW1 = (unsigned short*)alloc(128 * 128 * 2);
  unsigned short* lvW2 = (unsigned short*)alloc(64 * 128 * 2);
  unsigned short* emW1 = (unsigned short*)alloc(128 * 128 * 2);
  unsigned short* emW2 = (unsigned short*)alloc(128 * 128 * 2);
  unsigned short* roW1 = (unsigned short*)alloc(128 * 128 * 2);
  unsigned short* roW2 = (unsigned short*)alloc(32 * 128 * 2);
  prep(101, 128, 128, 128, 128, muW1); prep(103, 128, 64, 128, 64, muW2);
  prep(97,  128, 128, 128, 128, lvW1); prep(99,  128, 64, 128, 64, lvW2);
  prep(52,  64,  128, 128, 128, emW1); prep(54,  128, 128, 128, 128, emW2);
  prep(105, 128, 128, 128, 128, roW1); prep(107, 128, 1,   128, 32,  roW2);

  // ---- graph structure / counts ----
  build_edges_enc<<<(E1c + 255) / 256, 256, 0, stream>>>(eidx, batch, srcE, dstE, bsrcE, E1c);
  build_edges_dec<<<(EDHALF + 255) / 256, 256, 0, stream>>>(ae, dcei, batch, srcD, dstD, bsrcD, E3c, E2c);
  zero(nodeCnt,  (size_t)Bc * 4); count(batch, nodeCnt, N);
  zero(edgeCntE, (size_t)Bc * 4); count(bsrcE, edgeCntE, EENC);
  zero(edgeCntD, (size_t)Bc * 4); count(bsrcD, edgeCntD, EDEC);
  zero(inCntE,   (size_t)N * 4);  count(dstE, inCntE, EENC);
  zero(inCntD,   (size_t)N * 4);  count(dstD, inCntD, EDEC);

  // ---- one MPNN layer ----
  auto run_layer = [&](const MpnnWts& w, int Ecur, const int* src, const int* dst,
                       const int* bsrc, const float* inCnt, const float* edgeCnt) {
    // edge MLP: m = [h[src], h[dst], e, g[batch[src]]]  (K = 512)
    Chunks ce{};
    ce.base[0] = hB; ce.idx[0] = src;  ce.colOff[0] = 0; ce.rowStride[0] = 128;
    ce.base[1] = hB; ce.idx[1] = dst;  ce.colOff[1] = 0; ce.rowStride[1] = 128;
    ce.base[2] = eB; ce.idx[2] = nullptr; ce.colOff[2] = 0; ce.rowStride[2] = 128;
    ce.base[3] = gB; ce.idx[3] = bsrc; ce.colOff[3] = 0; ce.rowStride[3] = 128;
    gemm(ce, w.e1, w.e1b, Ecur, 512, 256, 256, 1, nullptr, 0, nullptr, mid, 256);
    Chunks cm{};
    cm.base[0] = mid; cm.idx[0] = nullptr; cm.colOff[0] = 0;   cm.rowStride[0] = 256;
    cm.base[1] = mid; cm.idx[1] = nullptr; cm.colOff[1] = 128; cm.rowStride[1] = 256;
    gemm(cm, w.e2, w.e2b, Ecur, 256, 128, 128, 0, eF, 128, tmpF, nullptr, 128);
    ln(tmpF, eF, eB, Ecur, 0);

    // agg = seg_mean(e_new, dst, N)
    zero(aggF, (size_t)N * 128 * 4);
    segsum(eF, dst, aggF, Ecur);
    segdiv(aggF, inCnt, nullptr, aggB, N);

    // node MLP: [h, agg, g[batch]]  (K = 384)
    Chunks cn{};
    cn.base[0] = hB;   cn.idx[0] = nullptr; cn.colOff[0] = 0; cn.rowStride[0] = 128;
    cn.base[1] = aggB; cn.idx[1] = nullptr; cn.colOff[1] = 0; cn.rowStride[1] = 128;
    cn.base[2] = gB;   cn.idx[2] = batch;   cn.colOff[2] = 0; cn.rowStride[2] = 128;
    gemm(cn, w.h1, w.h1b, N, 384, 256, 256, 1, nullptr, 0, nullptr, mid, 256);
    gemm(cm, w.h2, w.h2b, N, 256, 128, 128, 0, hF, 128, tmpF, nullptr, 128);
    ln(tmpF, hF, hB, N, 0);

    // gh = seg_mean(h_new, batch, B); gev = seg_mean(e_new, batch[src], B)
    zero(ghF, (size_t)Bc * 128 * 4);
    segsum(hF, batch, ghF, N);
    segdiv(ghF, nodeCnt, nullptr, ghB, Bc);
    zero(gevF, (size_t)Bc * 128 * 4);
    segsum(eF, bsrc, gevF, Ecur);
    segdiv(gevF, edgeCnt, nullptr, gevB, Bc);

    // graph MLP: [g, gh, gev]  (K = 384)
    Chunks cg{};
    cg.base[0] = gB;   cg.idx[0] = nullptr; cg.colOff[0] = 0; cg.rowStride[0] = 128;
    cg.base[1] = ghB;  cg.idx[1] = nullptr; cg.colOff[1] = 0; cg.rowStride[1] = 128;
    cg.base[2] = gevB; cg.idx[2] = nullptr; cg.colOff[2] = 0; cg.rowStride[2] = 128;
    gemm(cg, w.g1, w.g1b, Bc, 384, 256, 256, 1, nullptr, 0, nullptr, mid, 256);
    gemm(cm, w.g2, w.g2b, Bc, 256, 128, 128, 0, gF, 128, tmpF, nullptr, 128);
    ln(tmpF, gF, gB, Bc, 0);
  };

  // ======================= ENCODER =======================
  embed_kernel<<<(N * 128 + 255) / 256, 256, 0, stream>>>(
      hI, hjcI, hifI, hafI,
      (const float*)d_in[58], (const float*)d_in[60],
      (const float*)d_in[59], (const float*)d_in[56], hF, hB, N);
  edge_embed_kernel<<<(EENC * 128 + 255) / 256, 256, 0, stream>>>(
      ein, (const float*)d_in[57], eF, eB, E1c, 0);

  zero(gF, (size_t)Bc * 128 * 4);
  segsum(hF, batch, gF, N);
  segdiv(gF, nodeCnt, gF, gB, Bc);

  for (int l = 0; l < 3; ++l)
    run_layer(encW[l], EENC, srcE, dstE, bsrcE, inCntE, edgeCntE);

  // ======================= LATENT HEADS =======================
  Chunks cg1{};
  cg1.base[0] = gB; cg1.idx[0] = nullptr; cg1.colOff[0] = 0; cg1.rowStride[0] = 128;
  Chunks cm1{};
  cm1.base[0] = mid; cm1.idx[0] = nullptr; cm1.colOff[0] = 0; cm1.rowStride[0] = 128;

  gemm(cg1, muW1, (const float*)d_in[102], Bc, 128, 128, 128, 0, nullptr, 0, tmpF, nullptr, 128);
  ln(tmpF, nullptr, mid, Bc, 1);
  gemm(cm1, muW2, (const float*)d_in[104], Bc, 128, 64, 64, 0, nullptr, 0, out, nullptr, 64);

  gemm(cg1, lvW1, (const float*)d_in[98], Bc, 128, 128, 128, 0, nullptr, 0, tmpF, nullptr, 128);
  ln(tmpF, nullptr, mid, Bc, 1);
  gemm(cm1, lvW2, (const float*)d_in[100], Bc, 128, 64, 64, 0, nullptr, 0, out + BZ, nullptr, 64);

  zero(zB, (size_t)Bc * 128 * 2);
  z_kernel<<<(BZ + 255) / 256, 256, 0, stream>>>(eps, out, zB, BZ);

  Chunks cz{};
  cz.base[0] = zB; cz.idx[0] = nullptr; cz.colOff[0] = 0; cz.rowStride[0] = 128;
  gemm(cz, emW1, (const float*)d_in[53], Bc, 128, 128, 128, 0, nullptr, 0, tmpF, nullptr, 128);
  ln(tmpF, nullptr, mid, Bc, 1);
  gemm(cm1, emW2, (const float*)d_in[55], Bc, 128, 128, 128, 0, nullptr, 0, gF, gB, 128);

  // ======================= DECODER =======================
  embed_kernel<<<(N * 128 + 255) / 256, 256, 0, stream>>>(
      hI, hjcI, hifI, hafI,
      (const float*)d_in[13], (const float*)d_in[15],
      (const float*)d_in[14], (const float*)d_in[11], hF, hB, N);
  edge_embed_kernel<<<(EDEC * 128 + 255) / 256, 256, 0, stream>>>(
      dce, (const float*)d_in[12], eF, eB, EDHALF, E3c);

  for (int l = 0; l < 3; ++l)
    run_layer(decW[l], EDEC, srcD, dstD, bsrcD, inCntD, edgeCntD);

  // ======================= EDGE READOUT =======================
  Chunks ce1{};
  ce1.base[0] = eB; ce1.idx[0] = nullptr; ce1.colOff[0] = 0; ce1.rowStride[0] = 128;
  gemm(ce1, roW1, (const float*)d_in[106], E3c, 128, 128, 128, 0, nullptr, 0, tmpF, nullptr, 128);
  ln(tmpF, nullptr, mid, E3c, 1);
  Chunks cm2{};
  cm2.base[0] = mid; cm2.idx[0] = nullptr; cm2.colOff[0] = 0; cm2.rowStride[0] = 128;
  gemm(cm2, roW2, (const float*)d_in[108], E3c, 128, 32, 1, 0, nullptr, 0, out + 3 * BZ, nullptr, 1);
}
